// Agent_Aggregator_with_Mask_Denoise_Mechanism_45440753992237
// MI455X (gfx1250) — compile-verified, matrix operand order checked by reading, not run on hardware
//
#include <hip/hip_runtime.h>
#include <hip/hip_bf16.h>
#include <stdint.h>

// ---------- types ----------
typedef __attribute__((ext_vector_type(16))) __bf16   v16bf;
typedef __attribute__((ext_vector_type(8)))  float    v8f;
typedef __attribute__((ext_vector_type(8)))  uint32_t v8u;
typedef __attribute__((ext_vector_type(4)))  int      v4i;
typedef __attribute__((address_space(1))) v4i* gv4i_ptr;   // global int4*
typedef __attribute__((address_space(3))) v4i* lv4i_ptr;   // LDS int4*

#define B_  2
#define N_  8192
#define DIM_ 512
#define H_  8
#define A_  256
#define D_  64
#define NSPLIT 16   // n-dimension splits for the flash kv stage

// ---------- helpers ----------
__device__ __forceinline__ uint16_t f2bf(float f) {
    uint32_t u = __builtin_bit_cast(uint32_t, f);
    u += 0x7FFFu + ((u >> 16) & 1u);           // round-to-nearest-even
    return (uint16_t)(u >> 16);
}
__device__ __forceinline__ uint32_t pk2bf(float a, float b) {
    return (uint32_t)f2bf(a) | ((uint32_t)f2bf(b) << 16);
}

// ---- CDNA5 async global->LDS copy (ASYNCcnt path), with portable fallback ----
__device__ __forceinline__ void async_copy16(const void* gsrc, void* ldst) {
#if __has_builtin(__builtin_amdgcn_global_load_async_to_lds_b128)
    __builtin_amdgcn_global_load_async_to_lds_b128(
        (gv4i_ptr)(void*)gsrc, (lv4i_ptr)ldst, 0, 0);
#else
    *(uint4*)ldst = *(const uint4*)gsrc;
#endif
}
__device__ __forceinline__ void async_wait0() {
#if __has_builtin(__builtin_amdgcn_global_load_async_to_lds_b128)
#if __has_builtin(__builtin_amdgcn_s_wait_asynccnt)
    __builtin_amdgcn_s_wait_asynccnt(0);
#else
    asm volatile("s_wait_asynccnt 0x0" ::: "memory");
#endif
#endif
}

// WMMA bf16 fragment loader (ISA 7.12.2 layout): lane half (hi) holds two
// contiguous 8-element (16B) K-chunks: K in [hi*8, +8) and [16+hi*8, +8).
__device__ __forceinline__ v16bf load_frag(const uint32_t* p, int row, int strideU32,
                                           int hi, int kOffU32) {
    const uint32_t* base = p + (size_t)row * strideU32 + kOffU32 + hi * 4;
    uint4 c0 = *(const uint4*)(base);
    uint4 c1 = *(const uint4*)(base + 8);
    v8u u;
    u[0] = c0.x; u[1] = c0.y; u[2] = c0.z; u[3] = c0.w;
    u[4] = c1.x; u[5] = c1.y; u[6] = c1.z; u[7] = c1.w;
    return __builtin_bit_cast(v16bf, u);
}

__device__ __forceinline__ v8f wmma_bf16(v16bf a, v16bf b, v8f c) {
    return __builtin_amdgcn_wmma_f32_16x16x32_bf16(false, a, false, b, (short)0, c,
                                                   false, false);
}

__device__ __forceinline__ float rmax16(float v) {
    v = fmaxf(v, __shfl_xor(v, 1, 32));
    v = fmaxf(v, __shfl_xor(v, 2, 32));
    v = fmaxf(v, __shfl_xor(v, 4, 32));
    v = fmaxf(v, __shfl_xor(v, 8, 32));
    return v;
}
__device__ __forceinline__ float rsum16(float v) {
    v += __shfl_xor(v, 1, 32);
    v += __shfl_xor(v, 2, 32);
    v += __shfl_xor(v, 4, 32);
    v += __shfl_xor(v, 8, 32);
    return v;
}
__device__ __forceinline__ float sigmoidf(float x) { return 1.f / (1.f + __expf(-x)); }

// ---------------------------------------------------------------------------
// Kernel 0: f32 -> bf16 elementwise (agent tensor)
// ---------------------------------------------------------------------------
__global__ __launch_bounds__(256) void cvt_bf16_kernel(const float* __restrict__ in,
                                                       uint16_t* __restrict__ out, int n) {
    int i = blockIdx.x * 256 + threadIdx.x;
    if (i < n) out[i] = f2bf(in[i]);
}

// ---------------------------------------------------------------------------
// Kernel 1: qkv = x @ w_qkv.T  -> bf16 [16384][1536]
// ---------------------------------------------------------------------------
__global__ __launch_bounds__(256) void qkv_gemm_kernel(const float* __restrict__ X,
                                                       const float* __restrict__ W,
                                                       uint16_t* __restrict__ C) {
    __shared__ __align__(16) uint32_t ldsA[128 * 16];  // 128 rows x 32 bf16
    __shared__ __align__(16) uint32_t ldsB[128 * 16];

    const int m0 = blockIdx.y * 128;
    const int n0 = blockIdx.x * 128;
    const int t = threadIdx.x, lane = t & 31, wv = t >> 5;
    const int hi = lane >> 4, lr = lane & 15;
    const int wm = (wv & 3) * 32;   // 4 waves along M
    const int wn = (wv >> 2) * 64;  // 2 waves along N

    v8f acc[2][4] = {};

    for (int kk = 0; kk < DIM_; kk += 32) {
#pragma unroll
        for (int i = 0; i < 4; ++i) {
            int idx = t + i * 256;              // 0..1023
            int row = idx >> 3, c4 = idx & 7;   // 128 rows x 8 float4
            float4 xv = *(const float4*)(X + (size_t)(m0 + row) * DIM_ + kk + c4 * 4);
            ldsA[row * 16 + c4 * 2]     = pk2bf(xv.x, xv.y);
            ldsA[row * 16 + c4 * 2 + 1] = pk2bf(xv.z, xv.w);
            float4 wv4 = *(const float4*)(W + (size_t)(n0 + row) * DIM_ + kk + c4 * 4);
            ldsB[row * 16 + c4 * 2]     = pk2bf(wv4.x, wv4.y);
            ldsB[row * 16 + c4 * 2 + 1] = pk2bf(wv4.z, wv4.w);
            if (kk + 32 < DIM_)
                __builtin_prefetch(X + (size_t)(m0 + row) * DIM_ + kk + 32 + c4 * 4, 0, 0);
        }
        __syncthreads();

        v16bf af[2], bf[4];
#pragma unroll
        for (int i = 0; i < 2; ++i) af[i] = load_frag(ldsA, wm + i * 16 + lr, 16, hi, 0);
#pragma unroll
        for (int j = 0; j < 4; ++j) bf[j] = load_frag(ldsB, wn + j * 16 + lr, 16, hi, 0);
#pragma unroll
        for (int i = 0; i < 2; ++i)
#pragma unroll
            for (int j = 0; j < 4; ++j) acc[i][j] = wmma_bf16(af[i], bf[j], acc[i][j]);
        __syncthreads();
    }

#pragma unroll
    for (int i = 0; i < 2; ++i)
#pragma unroll
        for (int j = 0; j < 4; ++j) {
            int rbase = m0 + wm + i * 16 + hi * 8;
            int col = n0 + wn + j * 16 + lr;
#pragma unroll
            for (int r = 0; r < 8; ++r)
                C[(size_t)(rbase + r) * 1536 + col] = f2bf(acc[i][j][r]);
        }
}

// ---------------------------------------------------------------------------
// Kernel 2: flash partials for ka = softmax_n(agent @ k^T), kv = ka @ v
// Grid (bh, NSPLIT): each block handles n in [split*512, split*512+512).
// Outputs UNNORMALIZED acc + per-row running (max, sum).
// ---------------------------------------------------------------------------
__global__ __launch_bounds__(256) void kv_flash_partial_kernel(
        const uint16_t* __restrict__ qkvB, const uint16_t* __restrict__ agentB,
        float* __restrict__ kvPart, float* __restrict__ mPart, float* __restrict__ lPart) {
    const int bh = blockIdx.x, b = bh >> 3, h = bh & 7;
    const int split = blockIdx.y;
    __shared__ __align__(16) uint32_t Ag[256 * 32];   // agent[a=256][d=64] bf16
    __shared__ __align__(16) uint32_t Kt[32 * 32];    // k tile [n=32][d=64] bf16
    __shared__ __align__(16) uint32_t Vt[64 * 16];    // v tile transposed [d=64][n=32]
    __shared__ __align__(16) uint32_t Pt[8][32 * 16]; // per-wave p [a=32][n=32] bf16

    const int t = threadIdx.x, lane = t & 31, wv = t >> 5;
    const int hi = lane >> 4, lr = lane & 15;
    const int aRow = wv * 32;

    // stage agent via async global->LDS (8192 u32 = 2048 x 16B chunks)
    const uint32_t* agp = (const uint32_t*)agentB + (size_t)h * 256 * 32;
    for (int i = t; i < 2048; i += 256) async_copy16(agp + i * 4, Ag + i * 4);
    async_wait0();
    __syncthreads();

    v16bf aq[2][2];
#pragma unroll
    for (int rt = 0; rt < 2; ++rt)
#pragma unroll
        for (int kc = 0; kc < 2; ++kc)
            aq[rt][kc] = load_frag(Ag, aRow + rt * 16 + lr, 32, hi, kc * 16);

    v8f acc[2][4] = {};
    float mrun[2][8], lsum[2][8];
#pragma unroll
    for (int rt = 0; rt < 2; ++rt)
#pragma unroll
        for (int r = 0; r < 8; ++r) { mrun[rt][r] = -3.0e38f; lsum[rt][r] = 0.f; }

    const uint16_t* qkv16 = qkvB;
    const int nBeg = split * (N_ / NSPLIT);
    const int nEnd = nBeg + (N_ / NSPLIT);

    for (int n0 = nBeg; n0 < nEnd; n0 += 32) {
        __syncthreads();
        // stage k rows [n][d] via async: 32 rows x 128B = 256 chunks of 16B
        {
            int row = t >> 3, ch = t & 7;
            const void* src = qkv16 + (size_t)(b * N_ + n0 + row) * 1536 + 512 + h * 64 + ch * 8;
            async_copy16(src, Kt + row * 32 + ch * 4);
        }
        // stage v transposed (manual gather): Vt[d][n-pair]
        for (int i = t; i < 64 * 16; i += 256) {
            int d = i >> 4, np = i & 15;
            size_t r0 = (size_t)(b * N_ + n0 + 2 * np) * 1536 + 1024 + h * 64 + d;
            uint32_t lo = qkv16[r0];
            uint32_t hh = qkv16[r0 + 1536];
            Vt[i] = lo | (hh << 16);
        }
        async_wait0();
        __syncthreads();

        // s = agent @ k^T (no scale for ka per reference)
        v8f s[2][2] = {};
#pragma unroll
        for (int rt = 0; rt < 2; ++rt)
#pragma unroll
            for (int nt = 0; nt < 2; ++nt)
#pragma unroll
                for (int kc = 0; kc < 2; ++kc)
                    s[rt][nt] = wmma_bf16(aq[rt][kc],
                                          load_frag(Kt, nt * 16 + lr, 32, hi, kc * 16),
                                          s[rt][nt]);

        uint16_t* pw = (uint16_t*)Pt[wv];
#pragma unroll
        for (int rt = 0; rt < 2; ++rt)
#pragma unroll
            for (int r = 0; r < 8; ++r) {
                float s0 = s[rt][0][r], s1 = s[rt][1][r];
                float smax = rmax16(fmaxf(s0, s1));
                float nm = fmaxf(mrun[rt][r], smax);
                float fac = __expf(mrun[rt][r] - nm);
                float p0 = __expf(s0 - nm), p1 = __expf(s1 - nm);
                lsum[rt][r] = lsum[rt][r] * fac + rsum16(p0 + p1);
                mrun[rt][r] = nm;
#pragma unroll
                for (int dt = 0; dt < 4; ++dt) acc[rt][dt][r] *= fac;
                int row = rt * 16 + hi * 8 + r;
                pw[row * 32 + lr]      = f2bf(p0);
                pw[row * 32 + 16 + lr] = f2bf(p1);
            }

        v16bf vf[4];
#pragma unroll
        for (int dt = 0; dt < 4; ++dt) vf[dt] = load_frag(Vt, dt * 16 + lr, 16, hi, 0);
#pragma unroll
        for (int rt = 0; rt < 2; ++rt) {
            v16bf pf = load_frag((const uint32_t*)Pt[wv], rt * 16 + lr, 16, hi, 0);
#pragma unroll
            for (int dt = 0; dt < 4; ++dt) acc[rt][dt] = wmma_bf16(pf, vf[dt], acc[rt][dt]);
        }
    }

    // store unnormalized partials
    float* kp = kvPart + ((size_t)bh * NSPLIT + split) * A_ * D_;
    float* mp = mPart + ((size_t)bh * NSPLIT + split) * A_;
    float* lp = lPart + ((size_t)bh * NSPLIT + split) * A_;
#pragma unroll
    for (int rt = 0; rt < 2; ++rt)
#pragma unroll
        for (int r = 0; r < 8; ++r) {
            int row = aRow + rt * 16 + hi * 8 + r;
#pragma unroll
            for (int dt = 0; dt < 4; ++dt)
                kp[(size_t)row * 64 + dt * 16 + lr] = acc[rt][dt][r];
            if (lr == 0) { mp[row] = mrun[rt][r]; lp[row] = lsum[rt][r]; }
        }
}

// ---------------------------------------------------------------------------
// Kernel 2b: merge flash partials -> kv [bh][256][64] f32
// ---------------------------------------------------------------------------
__global__ __launch_bounds__(256) void kv_merge_kernel(const float* __restrict__ kvPart,
                                                       const float* __restrict__ mPart,
                                                       const float* __restrict__ lPart,
                                                       float* __restrict__ kvOut) {
    const int bh = blockIdx.x;
    const int a = threadIdx.x;   // 256 rows / 256 threads
    float M = -3.0e38f;
#pragma unroll
    for (int s = 0; s < NSPLIT; ++s)
        M = fmaxf(M, mPart[((size_t)bh * NSPLIT + s) * A_ + a]);
    float w[NSPLIT];
    float L = 0.f;
#pragma unroll
    for (int s = 0; s < NSPLIT; ++s) {
        w[s] = __expf(mPart[((size_t)bh * NSPLIT + s) * A_ + a] - M);
        L += lPart[((size_t)bh * NSPLIT + s) * A_ + a] * w[s];
    }
    float inv = 1.f / L;
    float o[64];
#pragma unroll
    for (int d = 0; d < 64; ++d) o[d] = 0.f;
    for (int s = 0; s < NSPLIT; ++s) {
        const float* kp = kvPart + (((size_t)bh * NSPLIT + s) * A_ + a) * (size_t)D_;
        float ws = w[s];
#pragma unroll
        for (int d = 0; d < 64; ++d) o[d] += ws * kp[d];
    }
    float* op = kvOut + ((size_t)bh * A_ + a) * (size_t)D_;
#pragma unroll
    for (int d = 0; d < 64; ++d) op[d] = o[d] * inv;
}

// ---------------------------------------------------------------------------
// Kernel 3: thresh = sigmoid(mean(kv_c @ w_thresh^T + b_thresh))
// ---------------------------------------------------------------------------
__global__ __launch_bounds__(256) void thresh_kernel(const float* __restrict__ kv,
                                                     const float* __restrict__ w_thresh,
                                                     const float* __restrict__ b_thresh,
                                                     float* __restrict__ thr) {
    __shared__ float red[256];
    const int t = threadIdx.x;
    float s = 0.f;
    for (int i = t; i < B_ * H_ * A_ * D_; i += 256) {
        int d = i & 63;
        int h = (i >> 14) & 7;   // i = ((b*8+h)*256 + a)*64 + d
        s += kv[i] * w_thresh[h * 64 + d];
    }
    red[t] = s;
    __syncthreads();
    for (int off = 128; off; off >>= 1) {
        if (t < off) red[t] += red[t + off];
        __syncthreads();
    }
    if (t == 0) {
        float mean = red[0] / (float)(B_ * A_) + b_thresh[0];
        thr[0] = sigmoidf(mean);
    }
}

// ---------------------------------------------------------------------------
// Kernel 4: per (b,h): denoise/mask/softmax_d -> kv2T[bh][d][a] bf16
// ---------------------------------------------------------------------------
__global__ __launch_bounds__(256) void kv_post_kernel(const float* __restrict__ kv,
                                                      const float* __restrict__ w_noise,
                                                      const float* __restrict__ b_noise,
                                                      const float* __restrict__ w_mask,
                                                      const float* __restrict__ b_mask,
                                                      const float* __restrict__ thr,
                                                      uint16_t* __restrict__ kv2T) {
    const int bh = blockIdx.x;
    __shared__ float W1[64 * 64];
    __shared__ float W2[64 * 64];
    const int t = threadIdx.x;
    for (int i = t; i < 64 * 64; i += 256) { W1[i] = w_noise[i]; W2[i] = w_mask[i]; }
    __syncthreads();

    const float th = thr[0];
    const int a = t;
    const float* kvr = kv + (size_t)bh * A_ * D_ + (size_t)a * D_;
    float row[64], vrow[64];
#pragma unroll
    for (int d = 0; d < 64; ++d) row[d] = kvr[d];

    float mx = -3.0e38f;
    for (int j = 0; j < 64; ++j) {
        float s1 = b_noise[j], s2 = b_mask[j];
#pragma unroll
        for (int d = 0; d < 64; ++d) {
            s1 += row[d] * W1[j * 64 + d];
            s2 += row[d] * W2[j * 64 + d];
        }
        float dn = sigmoidf(s1);
        float mk = sigmoidf(s2);
        float mbin = (mk > th) ? 1.f : 0.f;
        float val = row[j] * mbin + dn;
        vrow[j] = val;
        mx = fmaxf(mx, val);
    }
    float sum = 0.f;
#pragma unroll
    for (int j = 0; j < 64; ++j) { vrow[j] = __expf(vrow[j] - mx); sum += vrow[j]; }
    float inv = 1.f / sum;
    for (int j = 0; j < 64; ++j)
        kv2T[((size_t)bh * 64 + j) * 256 + a] = f2bf(vrow[j] * inv);
}

// ---------------------------------------------------------------------------
// Kernel 5: out = softmax_a(q @ agent^T * scale) @ kv2, fully fused
// ---------------------------------------------------------------------------
__global__ __launch_bounds__(256) void out_attn_kernel(const uint16_t* __restrict__ qkvB,
                                                       const uint16_t* __restrict__ agentB,
                                                       const uint16_t* __restrict__ kv2T,
                                                       float* __restrict__ out) {
    const int bh = blockIdx.x, b = bh >> 3, h = bh & 7;
    const int n0 = blockIdx.y * 128;
    __shared__ __align__(16) uint32_t KvT[64 * 128];  // kv2T [d=64][a=256] bf16
    __shared__ __align__(16) uint32_t Pt[8][16 * 16]; // per-wave p [16][32] bf16

    const int t = threadIdx.x, lane = t & 31, wv = t >> 5;
    const int hi = lane >> 4, lr = lane & 15;

    // stage kv2T via async global->LDS (8192 u32 = 2048 x 16B chunks)
    const uint32_t* kvsrc = (const uint32_t*)kv2T + (size_t)bh * 64 * 128;
    for (int i = t; i < 2048; i += 256) async_copy16(kvsrc + i * 4, KvT + i * 4);
    async_wait0();
    __syncthreads();

    const int m = n0 + wv * 16 + lr;
    const uint32_t* qrow = (const uint32_t*)qkvB + (size_t)(b * N_ + m) * 768 + h * 32;
    v16bf aq[2];
    aq[0] = load_frag(qrow, 0, 0, hi, 0);
    aq[1] = load_frag(qrow, 0, 0, hi, 16);

    const uint32_t* agp = (const uint32_t*)agentB + (size_t)h * 256 * 32;

    v8f acc[4] = {};
    float mrun[8], lsum[8];
#pragma unroll
    for (int r = 0; r < 8; ++r) { mrun[r] = -3.0e38f; lsum[r] = 0.f; }

    for (int a0 = 0; a0 < A_; a0 += 32) {
        v8f s[2] = {};
#pragma unroll
        for (int nt = 0; nt < 2; ++nt)
#pragma unroll
            for (int kc = 0; kc < 2; ++kc)
                s[nt] = wmma_bf16(aq[kc],
                                  load_frag(agp, a0 + nt * 16 + lr, 32, hi, kc * 16),
                                  s[nt]);

        uint16_t* pw = (uint16_t*)Pt[wv];
#pragma unroll
        for (int r = 0; r < 8; ++r) {
            float s0 = s[0][r] * 0.125f;   // scale = 64^-0.5
            float s1 = s[1][r] * 0.125f;
            float smax = rmax16(fmaxf(s0, s1));
            float nm = fmaxf(mrun[r], smax);
            float fac = __expf(mrun[r] - nm);
            float p0 = __expf(s0 - nm), p1 = __expf(s1 - nm);
            lsum[r] = lsum[r] * fac + rsum16(p0 + p1);
            mrun[r] = nm;
#pragma unroll
            for (int dt = 0; dt < 4; ++dt) acc[dt][r] *= fac;
            int row = hi * 8 + r;
            pw[row * 32 + lr]      = f2bf(p0);
            pw[row * 32 + 16 + lr] = f2bf(p1);
        }

        v16bf pf = load_frag((const uint32_t*)Pt[wv], lr, 16, hi, 0);
#pragma unroll
        for (int dt = 0; dt < 4; ++dt)
            acc[dt] = wmma_bf16(pf, load_frag(KvT, dt * 16 + lr, 128, hi, a0 >> 1),
                                acc[dt]);
    }

#pragma unroll
    for (int r = 0; r < 8; ++r) {
        float inv = 1.f / lsum[r];
        int row = n0 + wv * 16 + hi * 8 + r;
#pragma unroll
        for (int dt = 0; dt < 4; ++dt)
            out[((size_t)b * N_ + row) * 512 + h * 64 + dt * 16 + lr] = acc[dt][r] * inv;
    }
}

// ---------------------------------------------------------------------------
// launch
// ---------------------------------------------------------------------------
extern "C" void kernel_launch(void* const* d_in, const int* in_sizes, int n_in,
                              void* d_out, int out_size, void* d_ws, size_t ws_size,
                              hipStream_t stream) {
    const float* x        = (const float*)d_in[0];
    const float* w_qkv    = (const float*)d_in[1];
    const float* agent    = (const float*)d_in[2];
    const float* w_noise  = (const float*)d_in[3];
    const float* b_noise  = (const float*)d_in[4];
    const float* w_mask   = (const float*)d_in[5];
    const float* b_mask   = (const float*)d_in[6];
    const float* w_thresh = (const float*)d_in[7];
    const float* b_thresh = (const float*)d_in[8];
    float* out = (float*)d_out;

    char* ws = (char*)d_ws;
    uint16_t* qkvB   = (uint16_t*)(ws);                // 16384*1536 bf16 = 50331648 B
    uint16_t* agentB = (uint16_t*)(ws + 50331648);     // 131072 bf16     = 262144 B
    float*    kvPart = (float*)   (ws + 50593792);     // 16*16*256*64 f32 = 16777216 B
    float*    mPart  = (float*)   (ws + 67371008);     // 16*16*256 f32    = 262144 B
    float*    lPart  = (float*)   (ws + 67633152);     // 16*16*256 f32    = 262144 B
    float*    kvF    = (float*)   (ws + 67895296);     // 16*256*64 f32    = 1048576 B
    uint16_t* kv2T   = (uint16_t*)(ws + 68943872);     // 16*64*256 bf16   = 524288 B
    float*    thr    = (float*)   (ws + 69468160);     // 1 f32

    cvt_bf16_kernel<<<dim3((H_ * A_ * D_ + 255) / 256), dim3(256), 0, stream>>>(
        agent, agentB, H_ * A_ * D_);

    qkv_gemm_kernel<<<dim3(12, 128), dim3(256), 0, stream>>>(x, w_qkv, qkvB);

    kv_flash_partial_kernel<<<dim3(B_ * H_, NSPLIT), dim3(256), 0, stream>>>(
        qkvB, agentB, kvPart, mPart, lPart);

    kv_merge_kernel<<<dim3(B_ * H_), dim3(256), 0, stream>>>(kvPart, mPart, lPart, kvF);

    thresh_kernel<<<dim3(1), dim3(256), 0, stream>>>(kvF, w_thresh, b_thresh, thr);

    kv_post_kernel<<<dim3(B_ * H_), dim3(256), 0, stream>>>(kvF, w_noise, b_noise,
                                                            w_mask, b_mask, thr, kv2T);

    out_attn_kernel<<<dim3(B_ * H_, N_ / 128), dim3(256), 0, stream>>>(qkvB, agentB,
                                                                       kv2T, out);
}
